// TriangleSelfAttention_3109556322361
// MI455X (gfx1250) — matrix operand order repaired and checked by reading, not compiled, not run on hardware
//
#include <hip/hip_runtime.h>
#include <cmath>

// Triangle self-attention: B=1, N=256, C=128, H=4, Ch=32
#define SEQN 256
#define CDIM 128
#define NHEAD 4
#define CHEAD 32
#define LN_EPS 1e-5f

typedef __attribute__((ext_vector_type(16))) __bf16 v16bf;
typedef __attribute__((ext_vector_type(8)))  float  v8f;

// ---------------------------------------------------------------------------
// CDNA5 async LDS<->global copies (ASYNCcnt path).  The builtins take
// int4 pointers: param0 in the global (AS1/__device__) address space,
// param1 in the LDS (AS3/__shared__) address space, then imm offset + cpol.
// ---------------------------------------------------------------------------
#if __has_builtin(__builtin_amdgcn_global_load_async_to_lds_b128)
#define HAVE_ASYNC_LOAD 1
#else
#define HAVE_ASYNC_LOAD 0
#endif
#if __has_builtin(__builtin_amdgcn_global_store_async_from_lds_b128)
#define HAVE_ASYNC_STORE 1
#else
#define HAVE_ASYNC_STORE 0
#endif

typedef int v4i_vec __attribute__((vector_size(16)));
typedef __attribute__((address_space(1))) v4i_vec g_v4i;   // global int4
typedef __attribute__((address_space(3))) v4i_vec l_v4i;   // LDS int4

__device__ __forceinline__ void g2l_b128(__bf16* lds_dst, const __bf16* gsrc) {
#if HAVE_ASYNC_LOAD
  __builtin_amdgcn_global_load_async_to_lds_b128(
      (g_v4i*)gsrc, (l_v4i*)lds_dst, 0, 0);
#else
  *(uint4*)lds_dst = *(const uint4*)gsrc;
#endif
}

__device__ __forceinline__ void l2g_b128(__bf16* gdst, const __bf16* lds_src) {
#if HAVE_ASYNC_STORE
  __builtin_amdgcn_global_store_async_from_lds_b128(
      (g_v4i*)gdst, (l_v4i*)lds_src, 0, 0);
#else
  *(uint4*)gdst = *(const uint4*)lds_src;
#endif
}

__device__ __forceinline__ void wait_async() {
#if HAVE_ASYNC_LOAD || HAVE_ASYNC_STORE
#if __has_builtin(__builtin_amdgcn_s_wait_asynccnt)
  __builtin_amdgcn_s_wait_asynccnt(0);
#else
  asm volatile("s_wait_asynccnt 0" ::: "memory");
#endif
#endif
}

// ---------------------------------------------------------------------------
// WMMA fragment helpers (wave32; A and B share contiguous k = 16*(lane/16) + e)
// ---------------------------------------------------------------------------

// A-matrix 16x32 bf16 from a row-major tile (stride in elements).
__device__ __forceinline__ v16bf wmma_ldA_rowmaj(const __bf16* base, int stride, int kbase) {
  int lane = threadIdx.x & 31;
  int m = lane & 15, hi = lane >> 4;
  const __bf16* p = base + m * stride + kbase + hi * 16;
  v16bf a;
#pragma unroll
  for (int e = 0; e < 16; ++e) a[e] = p[e];
  return a;
}

// B-matrix 32x16 bf16, source laid out src[n][k] (K contiguous per N-row).
__device__ __forceinline__ v16bf wmma_ldB_kcontig(const __bf16* base, int stride) {
  int lane = threadIdx.x & 31;
  int n = lane & 15, hi = lane >> 4;
  const __bf16* p = base + n * stride + hi * 16;
  v16bf b;
#pragma unroll
  for (int e = 0; e < 16; ++e) b[e] = p[e];
  return b;
}

// B-matrix 32x16 bf16, source [k][n] (strided gather; V staged in LDS).
__device__ __forceinline__ v16bf wmma_ldB_kmajor(const __bf16* base, int stride) {
  int lane = threadIdx.x & 31;
  int n = lane & 15, hi = lane >> 4;
  const __bf16* p = base + (hi * 16) * stride + n;
  v16bf b;
#pragma unroll
  for (int e = 0; e < 16; ++e) b[e] = p[e * stride];
  return b;
}

__device__ __forceinline__ v8f wmma_bf16(v16bf a, v16bf b, v8f c) {
  return __builtin_amdgcn_wmma_f32_16x16x32_bf16(false, a, false, b, (short)0, c,
                                                 false, false);
}

__device__ __forceinline__ float xormax16(float v) {
  v = fmaxf(v, __shfl_xor(v, 1, 32));
  v = fmaxf(v, __shfl_xor(v, 2, 32));
  v = fmaxf(v, __shfl_xor(v, 4, 32));
  v = fmaxf(v, __shfl_xor(v, 8, 32));
  return v;
}
__device__ __forceinline__ float xorsum16(float v) {
  v += __shfl_xor(v, 1, 32);
  v += __shfl_xor(v, 2, 32);
  v += __shfl_xor(v, 4, 32);
  v += __shfl_xor(v, 8, 32);
  return v;
}

// ---------------------------------------------------------------------------
// Kernel 1: convert the 5 weight matrices (f32 [n][k]) to bf16, same layout.
// ---------------------------------------------------------------------------
__global__ void convert_weights(const float* Wq, const float* Wk, const float* Wv,
                                const float* Wg, const float* Wo, __bf16* Wb) {
  int idx = blockIdx.x * 256 + threadIdx.x;   // 5 * 128*128 total
  const float* Ws[5] = {Wq, Wk, Wv, Wg, Wo};
  int w = idx >> 14;
  int e = idx & 16383;
  if (w < 5) Wb[idx] = (__bf16)Ws[w][e];
}

// ---------------------------------------------------------------------------
// Kernel 2: fused Q/K/V/Gate projection.  Block = 256 threads = 8 waves,
// 128 rows per block.  pair tile staged to LDS as bf16; A-fragments held in
// registers so the LDS tile can be reused as an epilogue staging buffer that
// is drained with coalesced 16-byte async stores.
// ---------------------------------------------------------------------------
__launch_bounds__(256, 1)
__global__ void qkvg_proj(const float* x, const __bf16* Wb,
                          const float* bq, const float* bk, const float* bv,
                          const float* bg,
                          __bf16* Q, __bf16* K, __bf16* V, __bf16* G) {
  __shared__ __align__(16) __bf16 xs[128 * 128];   // 32 KB
  const long rowbase = (long)blockIdx.x * 128;
  for (int idx = threadIdx.x; idx < 128 * 128; idx += 256) {
    __builtin_prefetch(x + rowbase * CDIM + idx + 8192, 0, 1);
    xs[idx] = (__bf16)x[rowbase * CDIM + idx];
  }
  __syncthreads();

  const int wave = threadIdx.x >> 5;
  const int lane = threadIdx.x & 31, n = lane & 15, hi = lane >> 4;
  const int mbase = wave * 16;

  v16bf a[4];
#pragma unroll
  for (int kt = 0; kt < 4; ++kt)
    a[kt] = wmma_ldA_rowmaj(xs + mbase * 128, 128, kt * 32);
  __syncthreads();   // all waves have their A-fragments; xs is reusable now

  const float* bias[4] = {bq, bk, bv, bg};
  __bf16* outb[4] = {Q, K, V, G};

#pragma unroll
  for (int w = 0; w < 4; ++w) {
#pragma unroll
    for (int nt = 0; nt < 8; ++nt) {
      v8f acc = {};
#pragma unroll
      for (int kt = 0; kt < 4; ++kt) {
        v16bf b = wmma_ldB_kcontig(Wb + w * 16384 + (nt * 16) * 128 + kt * 32, 128);
        acc = wmma_bf16(a[kt], b, acc);
      }
      const int col = nt * 16 + n;
      const float bval = bias[w][col];
#pragma unroll
      for (int r = 0; r < 8; ++r) {
        float vv = acc[r] + bval;
        if (w == 3) vv = 1.f / (1.f + __expf(-vv));   // sigmoid gate
        xs[(mbase + r + hi * 8) * CDIM + col] = (__bf16)vv;
      }
    }
    __syncthreads();
    // Drain the full 128x128 bf16 tile (contiguous 32 KB in global).
    __bf16* dst = outb[w] + rowbase * CDIM;
    for (int q = threadIdx.x; q < 2048; q += 256)
      l2g_b128(dst + q * 8, xs + q * 8);
    wait_async();
    __syncthreads();
  }
}

// ---------------------------------------------------------------------------
// Kernel 3: per-row attention.  grid = (jhalf=2, i=256, h=4); block = 8 waves,
// each wave owns a 16-row j-tile.  S = Q*K^T is one WMMA per 16x16 tile
// (K-dim = Ch = 32); softmax via shfl-xor; P restaged through a per-wave LDS
// buffer into A-fragments for P*V; gated output drained with b128 stores.
// ---------------------------------------------------------------------------
__launch_bounds__(256, 1)
__global__ void attention(const __bf16* Q, const __bf16* K, const __bf16* V,
                          const __bf16* G, __bf16* AG) {
  __shared__ __align__(16) __bf16 Vs[SEQN * CHEAD];  // 16 KB : V[k][c] for (i,h)
  __shared__ __align__(16) __bf16 Ps[8 * 16 * 32];   //  8 KB : per-wave staging
  const int jblk = blockIdx.x;               // 0..1
  const int i = blockIdx.y;                  // 0..255
  const int h = blockIdx.z;                  // 0..3
  const long rbase = (long)i * SEQN;         // row base into [65536][128]

  // Stage V with async 16-byte LDS loads: chunk q covers (k=q>>2, c=(q&3)*8).
  for (int q = threadIdx.x; q < SEQN * CHEAD / 8; q += 256)
    g2l_b128(Vs + q * 8, V + (rbase + (q >> 2)) * CDIM + h * CHEAD + (q & 3) * 8);
  wait_async();
  __syncthreads();

  const int wave = threadIdx.x >> 5;
  const int lane = threadIdx.x & 31, n = lane & 15, hi = lane >> 4;
  const int jbase = jblk * 128 + wave * 16;

  // Q A-fragment straight from global (k-contiguous per row).
  v16bf aq;
  {
    const __bf16* p = Q + (rbase + jbase + n) * CDIM + h * CHEAD + hi * 16;
#pragma unroll
    for (int e = 0; e < 16; ++e) aq[e] = p[e];
  }

  // Logits: 16 k-tiles of 16x16, one WMMA each.
  v8f s[16];
#pragma unroll
  for (int kt = 0; kt < 16; ++kt) {
    v16bf bk = wmma_ldB_kcontig(K + (rbase + kt * 16) * CDIM + h * CHEAD, CDIM);
    v8f z = {};
    s[kt] = wmma_bf16(aq, bk, z);
  }

  const float scale = 0.17677669529663687f;  // Ch^-0.5
  float mx[8], sm[8];
#pragma unroll
  for (int r = 0; r < 8; ++r) {
    float m = -1e30f;
#pragma unroll
    for (int kt = 0; kt < 16; ++kt) m = fmaxf(m, s[kt][r]);
    mx[r] = xormax16(m);
  }
#pragma unroll
  for (int r = 0; r < 8; ++r) {
    float ss = 0.f;
#pragma unroll
    for (int kt = 0; kt < 16; ++kt) {
      float p = __expf((s[kt][r] - mx[r]) * scale);
      s[kt][r] = p;
      ss += p;
    }
    sm[r] = xorsum16(ss);
  }

  // O = P @ V in 32-wide K chunks; P goes through per-wave LDS to convert the
  // C/D register layout into an A-fragment layout.
  v8f o0 = {}, o1 = {};
  __bf16* myP = Ps + wave * (16 * 32);
#pragma unroll
  for (int kc = 0; kc < 8; ++kc) {
#pragma unroll
    for (int half = 0; half < 2; ++half) {
      const int kt = kc * 2 + half;
#pragma unroll
      for (int r = 0; r < 8; ++r) {
        int M = r + hi * 8;
        myP[M * 32 + half * 16 + n] = (__bf16)s[kt][r];
      }
    }
    asm volatile("s_wait_dscnt 0" ::: "memory");   // order LDS write -> read
    v16bf ap = wmma_ldA_rowmaj(myP, 32, 0);
    v16bf b0 = wmma_ldB_kmajor(Vs + (kc * 32) * CHEAD + 0, CHEAD);
    v16bf b1 = wmma_ldB_kmajor(Vs + (kc * 32) * CHEAD + 16, CHEAD);
    o0 = wmma_bf16(ap, b0, o0);
    o1 = wmma_bf16(ap, b1, o1);
  }

  // Epilogue: scale by 1/rowsum, gate, stage to per-wave LDS, drain with
  // coalesced 16-byte stores (wave's 16x32 tile = 64 chunks, 2 per lane).
#pragma unroll
  for (int r = 0; r < 8; ++r) {
    const int M = r + hi * 8;
    const long row = rbase + jbase + M;
    const float inv = 1.f / sm[r];
    int col = h * CHEAD + n;
    float g = (float)G[row * CDIM + col];
    myP[M * 32 + n] = (__bf16)(o0[r] * inv * g);
    col = h * CHEAD + 16 + n;
    g = (float)G[row * CDIM + col];
    myP[M * 32 + 16 + n] = (__bf16)(o1[r] * inv * g);
  }
  asm volatile("s_wait_dscnt 0" ::: "memory");
#pragma unroll
  for (int j = 0; j < 2; ++j) {
    const int q = lane + 32 * j;               // 0..63
    const int row = q >> 2, c8 = (q & 3) * 8;
    l2g_b128(AG + (rbase + jbase + row) * CDIM + h * CHEAD + c8, myP + q * 8);
  }
}

// ---------------------------------------------------------------------------
// Kernel 4: output projection + bias + residual + LayerNorm (f32 out).
// ---------------------------------------------------------------------------
__launch_bounds__(256, 1)
__global__ void out_ln(const __bf16* AG, const __bf16* Wo_bf, const float* bo,
                       const float* pair, const float* gamma, const float* beta,
                       float* out) {
  __shared__ __align__(16) __bf16 xs[128 * 128];   // 32 KB
  const long rowbase = (long)blockIdx.x * 128;
  // Async-stage the contiguous 32 KB AG tile into LDS.
  for (int q = threadIdx.x; q < 2048; q += 256)
    g2l_b128(xs + q * 8, AG + rowbase * CDIM + q * 8);
  wait_async();
  __syncthreads();

  const int wave = threadIdx.x >> 5;
  const int lane = threadIdx.x & 31, n = lane & 15, hi = lane >> 4;
  const int mbase = wave * 16;

  v16bf a[4];
#pragma unroll
  for (int kt = 0; kt < 4; ++kt)
    a[kt] = wmma_ldA_rowmaj(xs + mbase * 128, 128, kt * 32);

  float res[8][8];   // [nt][r]
#pragma unroll
  for (int nt = 0; nt < 8; ++nt) {
    v8f acc = {};
#pragma unroll
    for (int kt = 0; kt < 4; ++kt) {
      v16bf b = wmma_ldB_kcontig(Wo_bf + (nt * 16) * 128 + kt * 32, 128);
      acc = wmma_bf16(a[kt], b, acc);
    }
    const int col = nt * 16 + n;
    const float bias = bo[col];
#pragma unroll
    for (int r = 0; r < 8; ++r) {
      long row = rowbase + mbase + r + hi * 8;
      res[nt][r] = pair[row * CDIM + col] + acc[r] + bias;
    }
  }

#pragma unroll
  for (int r = 0; r < 8; ++r) {
    float s1 = 0.f, s2 = 0.f;
#pragma unroll
    for (int nt = 0; nt < 8; ++nt) { float v = res[nt][r]; s1 += v; s2 += v * v; }
    s1 = xorsum16(s1);
    s2 = xorsum16(s2);
    const float mu = s1 * (1.f / 128.f);
    const float var = s2 * (1.f / 128.f) - mu * mu;
    const float rstd = rsqrtf(var + LN_EPS);
    const long row = rowbase + mbase + r + hi * 8;
#pragma unroll
    for (int nt = 0; nt < 8; ++nt) {
      const int col = nt * 16 + n;
      out[row * CDIM + col] = (res[nt][r] - mu) * rstd * gamma[col] + beta[col];
    }
  }
}

// ---------------------------------------------------------------------------
extern "C" void kernel_launch(void* const* d_in, const int* in_sizes, int n_in,
                              void* d_out, int out_size, void* d_ws, size_t ws_size,
                              hipStream_t stream) {
  const float* pair = (const float*)d_in[0];
  const float* Wq = (const float*)d_in[1];  const float* bq = (const float*)d_in[2];
  const float* Wk = (const float*)d_in[3];  const float* bk = (const float*)d_in[4];
  const float* Wv = (const float*)d_in[5];  const float* bv = (const float*)d_in[6];
  const float* Wg = (const float*)d_in[7];  const float* bg = (const float*)d_in[8];
  const float* Wo = (const float*)d_in[9];  const float* bo = (const float*)d_in[10];
  const float* gamma = (const float*)d_in[11];
  const float* beta  = (const float*)d_in[12];

  char* ws = (char*)d_ws;
  __bf16* Wb = (__bf16*)ws;                        // 5 * 128*128 bf16
  size_t off = (size_t)5 * 16384 * sizeof(__bf16);
  off = (off + 255) & ~(size_t)255;
  const size_t MAT = (size_t)65536 * CDIM * sizeof(__bf16);  // 16 MB each
  __bf16* Qb = (__bf16*)(ws + off); off += MAT;
  __bf16* Kb = (__bf16*)(ws + off); off += MAT;
  __bf16* Vb = (__bf16*)(ws + off); off += MAT;
  __bf16* Gb = (__bf16*)(ws + off); off += MAT;
  __bf16* AG = (__bf16*)(ws + off); off += MAT;

  convert_weights<<<320, 256, 0, stream>>>(Wq, Wk, Wv, Wg, Wo, Wb);
  qkvg_proj<<<512, 256, 0, stream>>>(pair, Wb, bq, bk, bv, bg, Qb, Kb, Vb, Gb);
  attention<<<dim3(2, SEQN, NHEAD), 256, 0, stream>>>(Qb, Kb, Vb, Gb, AG);
  out_ln<<<512, 256, 0, stream>>>(AG, Wb + 4 * 16384, bo, pair, gamma, beta,
                                  (float*)d_out);
}
